// gcl_IS_79663053406799
// MI455X (gfx1250) — compile-verified, matrix operand order checked by reading, not run on hardware
//
#include <hip/hip_runtime.h>
#include <hip/hip_bf16.h>

// ---------------------------------------------------------------------------
// Contrastive loss (NT-Xent-like): N=8192, D=1024, TAU=1, ALPHA=1, BETA=0
// out[0] = loss.mean(), out[1] = p_ij.mean()
// Fused f16-WMMA GEMM (matrix never materialized, 32MB operands live in L2),
// double-buffered LDS tiles filled by async global->LDS copies when available.
// ---------------------------------------------------------------------------

typedef __attribute__((ext_vector_type(16))) _Float16 v16h;
typedef __attribute__((ext_vector_type(8)))  _Float16 v8h;
typedef __attribute__((ext_vector_type(4)))  _Float16 v4h;
typedef __attribute__((ext_vector_type(8)))  float    v8f;
typedef int v4i __attribute__((vector_size(16)));   // matches builtin param type

#define NROWS   8192
#define DIM     1024
#define TOTROWS 16384

// ---- gfx1250 async global->LDS copy (guarded; falls back to staged copy) ---
#if defined(__has_builtin)
#  if __has_builtin(__builtin_amdgcn_global_load_async_to_lds_b128)
#    define HAVE_ASYNC_LDS 1
#  endif
#endif
#ifndef HAVE_ASYNC_LDS
#  define HAVE_ASYNC_LDS 0
#endif

#if HAVE_ASYNC_LDS
#define GLB1 __attribute__((address_space(1)))
#define LDS3 __attribute__((address_space(3)))
__device__ __forceinline__ void async_b128(const void* g, const void* l) {
    // AS1 src (inttoptr keeps the 64-bit flat address), AS3 dst (low 32 bits
    // of a flat LDS address are the LDS byte offset).
    __builtin_amdgcn_global_load_async_to_lds_b128(
        (GLB1 v4i*)(unsigned long long)g,
        (LDS3 v4i*)(unsigned)(unsigned long long)l,
        0, 0);
}
__device__ __forceinline__ void wait_async0() {
#  if __has_builtin(__builtin_amdgcn_s_wait_asynccnt)
    __builtin_amdgcn_s_wait_asynccnt(0);
#  else
    asm volatile("s_wait_asynccnt 0x0" ::: "memory");
#  endif
}
#else
__device__ __forceinline__ void wait_async0() {}
#endif

// ---------------------------------------------------------------------------
// Kernel 0: zero scratch accumulators
// ---------------------------------------------------------------------------
__global__ void k_zero(float* p, int n) {
    int i = blockIdx.x * blockDim.x + threadIdx.x;
    if (i < n) p[i] = 0.0f;
}

// ---------------------------------------------------------------------------
// Kernel 1: per-row L2 normalize fp32 z -> f16 y  (one block per row)
// ---------------------------------------------------------------------------
__global__ __launch_bounds__(256) void k_rownorm(const float* __restrict__ z,
                                                 _Float16* __restrict__ y) {
    const int row  = blockIdx.x;      // 0..16383
    const int tid  = threadIdx.x;     // 0..255, 4 elems each
    const int lane = tid & 31;
    const int wid  = tid >> 5;

    const float4 v = *(const float4*)(z + (size_t)row * DIM + tid * 4);
    float ss = v.x * v.x + v.y * v.y + v.z * v.z + v.w * v.w;

    #pragma unroll
    for (int off = 1; off < 32; off <<= 1) ss += __shfl_xor(ss, off, 32);

    __shared__ float sred[8];
    __shared__ float stot;
    if (lane == 0) sred[wid] = ss;
    __syncthreads();
    if (tid < 8) {
        float t = sred[tid];
        #pragma unroll
        for (int off = 1; off < 8; off <<= 1) t += __shfl_xor(t, off, 8);
        if (tid == 0) stot = t;
    }
    __syncthreads();

    const float rn = rsqrtf(stot);
    v4h o;
    o.x = (_Float16)(v.x * rn);
    o.y = (_Float16)(v.y * rn);
    o.z = (_Float16)(v.z * rn);
    o.w = (_Float16)(v.w * rn);
    *(v4h*)(y + (size_t)row * DIM + tid * 4) = o;
}

// ---------------------------------------------------------------------------
// Kernel 2: fused 256x128-tile WMMA GEMM + exp/row-sum epilogue.
// 256 threads = 8 waves, 4 (M) x 2 (N); each wave owns 64x64 = 4x4 WMMA tiles
// -> 16 v_wmma_f32_16x16x32_f16 per 16 ds_load_b128 per K-step of 32.
// LDS tiles double-buffered; filled by async global->LDS DMA when available.
// ---------------------------------------------------------------------------
#define APAD 40   // padded LDS row stride in halves (80B -> conflict-free b128)
#define BM   256
#define BN   128
#define NK   (DIM / 32)

__global__ __launch_bounds__(256) void k_gemm(const _Float16* __restrict__ y,
                                              float* __restrict__ g_rs_exp,
                                              float* __restrict__ g_rs_m,
                                              float* __restrict__ g_possim) {
    __shared__ __align__(16) _Float16 Abuf[2][BM * APAD];
    __shared__ __align__(16) _Float16 Bbuf[2][BN * APAD];
    __shared__ float lds_e[BM];
    __shared__ float lds_m[BM];

    const int tid  = threadIdx.x;
    const int lane = tid & 31;
    const int w    = tid >> 5;        // wave 0..7
    const int wm   = w & 3;           // 64-row group
    const int wn   = w >> 2;          // 64-col group
    const int half = lane >> 4;       // 0|1
    const int mrow = lane & 15;

    const int br = blockIdx.y;        // row tile (x1), 256 rows
    const int bc = blockIdx.x;        // col tile (x2), 128 cols

    const _Float16* y1 = y;                               // rows of x1
    const _Float16* y2 = y + (size_t)NROWS * DIM;         // rows of x2

    if (tid < BM) { lds_e[tid] = 0.0f; lds_m[tid] = 0.0f; }

    // ---- tile copy: A = 256x32 halves (1024 16B chunks), B = 128x32 (512) --
    auto copy_tiles = [&](int buf, int kb) {
        #pragma unroll
        for (int i = 0; i < 4; ++i) {                    // A chunks
            const int c  = tid + i * 256;
            const int r  = c >> 2;
            const int ko = (c & 3) << 3;
            const void* g = y1 + (size_t)(br * BM + r) * DIM + kb + ko;
            void*       l = &Abuf[buf][r * APAD + ko];
#if HAVE_ASYNC_LDS
            async_b128(g, l);
#else
            *(uint4*)l = *(const uint4*)g;
#endif
        }
        #pragma unroll
        for (int i = 0; i < 2; ++i) {                    // B chunks
            const int c  = tid + i * 256;
            const int r  = c >> 2;
            const int ko = (c & 3) << 3;
            const void* g = y2 + (size_t)(bc * BN + r) * DIM + kb + ko;
            void*       l = &Bbuf[buf][r * APAD + ko];
#if HAVE_ASYNC_LDS
            async_b128(g, l);
#else
            *(uint4*)l = *(const uint4*)g;
#endif
        }
    };

    v8f acc[4][4] = {};

    copy_tiles(0, 0);                                    // prologue

    for (int it = 0; it < NK; ++it) {
        const int cur = it & 1;
        wait_async0();
        __syncthreads();                                 // buf cur ready; buf cur^1 free
        if (it + 1 < NK) copy_tiles(cur ^ 1, (it + 1) * 32);

        // ---- fragments (CDNA5 16-bit layout: v0-3 K=half*8.., v4-7 +16) ----
        v16h bf[4];
        #pragma unroll
        for (int nt = 0; nt < 4; ++nt) {
            const int r = wn * 64 + nt * 16 + mrow;
            v8h lo = *(const v8h*)(&Bbuf[cur][r * APAD + half * 8]);
            v8h hi = *(const v8h*)(&Bbuf[cur][r * APAD + 16 + half * 8]);
            bf[nt] = __builtin_shufflevector(lo, hi, 0,1,2,3,4,5,6,7,
                                                     8,9,10,11,12,13,14,15);
        }
        #pragma unroll
        for (int mt = 0; mt < 4; ++mt) {
            const int r = wm * 64 + mt * 16 + mrow;
            v8h lo = *(const v8h*)(&Abuf[cur][r * APAD + half * 8]);
            v8h hi = *(const v8h*)(&Abuf[cur][r * APAD + 16 + half * 8]);
            v16h af = __builtin_shufflevector(lo, hi, 0,1,2,3,4,5,6,7,
                                                      8,9,10,11,12,13,14,15);
            #pragma unroll
            for (int nt = 0; nt < 4; ++nt)
                acc[mt][nt] = __builtin_amdgcn_wmma_f32_16x16x32_f16(
                    false, af, false, bf[nt],
                    (short)0, acc[mt][nt], false, false);
        }
    }

    // ---- epilogue: exp, diagonal, row reductions ----
    #pragma unroll
    for (int mt = 0; mt < 4; ++mt) {
        #pragma unroll
        for (int r = 0; r < 8; ++r) {
            const int m_local = wm * 64 + mt * 16 + r + half * 8;
            const int grow    = br * BM + m_local;
            float sm = 0.0f, se = 0.0f;
            #pragma unroll
            for (int nt = 0; nt < 4; ++nt) {
                const float val = acc[mt][nt][r];     // TAU == 1
                const int gcol  = bc * BN + wn * 64 + nt * 16 + mrow;
                sm += val;
                se += __expf(val);
                if (grow == gcol) g_possim[grow] = val;
            }
            // reduce the 16 column-lanes (xor<16 stays within each half-wave)
            #pragma unroll
            for (int off = 1; off < 16; off <<= 1) {
                sm += __shfl_xor(sm, off, 32);
                se += __shfl_xor(se, off, 32);
            }
            if (mrow == 0) {
                atomicAdd(&lds_m[m_local], sm);   // ds_add_f32
                atomicAdd(&lds_e[m_local], se);
            }
        }
    }
    __syncthreads();
    if (tid < BM) {
        atomicAdd(&g_rs_m[br * BM + tid], lds_m[tid]);   // global_atomic_add_f32
        atomicAdd(&g_rs_exp[br * BM + tid], lds_e[tid]);
    }
}

// ---------------------------------------------------------------------------
// Kernel 3: finalize -> d_out[0] = mean loss, d_out[1] = mean p_ij
// loss_i = -log(p) - p * pos_sim   (alpha = p_ij - 1 + 1 = p_ij; BETA = 0)
// ---------------------------------------------------------------------------
__global__ __launch_bounds__(256) void k_finalize(const float* __restrict__ rs_e,
                                                  const float* __restrict__ rs_m,
                                                  const float* __restrict__ ps,
                                                  float* __restrict__ out) {
    const int tid  = threadIdx.x;
    const int lane = tid & 31;
    const int wid  = tid >> 5;

    float al = 0.0f, ap = 0.0f;
    for (int i = tid; i < NROWS; i += 256) {
        const float pos = ps[i];
        const float se  = rs_e[i];
        const float ep  = __expf(pos);
        const float p   = ep / (se - ep);
        al += -__logf(p) - p * pos;       // BETA==0 -> neg_sim term drops
        ap += p;
        (void)rs_m;                        // rs_m only needed if BETA != 0
    }
    #pragma unroll
    for (int off = 1; off < 32; off <<= 1) {
        al += __shfl_xor(al, off, 32);
        ap += __shfl_xor(ap, off, 32);
    }
    __shared__ float sl[8], sp[8];
    if (lane == 0) { sl[wid] = al; sp[wid] = ap; }
    __syncthreads();
    if (tid < 8) {
        float l = sl[tid], p = sp[tid];
        #pragma unroll
        for (int off = 1; off < 8; off <<= 1) {
            l += __shfl_xor(l, off, 8);
            p += __shfl_xor(p, off, 8);
        }
        if (tid == 0) {
            out[0] = l / (float)NROWS;
            out[1] = p / (float)NROWS;
        }
    }
}

// ---------------------------------------------------------------------------
extern "C" void kernel_launch(void* const* d_in, const int* in_sizes, int n_in,
                              void* d_out, int out_size, void* d_ws, size_t ws_size,
                              hipStream_t stream) {
    (void)in_sizes; (void)n_in; (void)out_size; (void)ws_size;
    const float* z   = (const float*)d_in[0];
    float*       out = (float*)d_out;

    char* ws = (char*)d_ws;
    _Float16* yv   = (_Float16*)ws;                                   // 32 MB f16
    float*    rs_e = (float*)(ws + (size_t)TOTROWS * DIM * sizeof(_Float16));
    float*    rs_m = rs_e + NROWS;
    float*    ps   = rs_m + NROWS;

    k_zero<<<(3 * NROWS + 255) / 256, 256, 0, stream>>>(rs_e, 3 * NROWS);
    k_rownorm<<<TOTROWS, 256, 0, stream>>>(z, yv);
    k_gemm<<<dim3(NROWS / BN, NROWS / BM), 256, 0, stream>>>(yv, rs_e, rs_m, ps);
    k_finalize<<<1, 256, 0, stream>>>(rs_e, rs_m, ps, out);
}
